// Net_11390253269720
// MI455X (gfx1250) — matrix operand order repaired and checked by reading, not compiled, not run on hardware
//
#include <hip/hip_runtime.h>
#include <stdint.h>

// CDNA5 / gfx1250, wave32.
typedef __attribute__((ext_vector_type(16))) _Float16 v16h;
typedef __attribute__((ext_vector_type(8)))  float    v8f;

constexpr int F = 32;  // feature width everywhere in this net

// ---------------------------------------------------------------------------
// K0: zero the degree counters (workspace is poisoned; must re-init per call)
// ---------------------------------------------------------------------------
__global__ void zero_deg_kernel(unsigned* __restrict__ deg, int n) {
  int i = blockIdx.x * blockDim.x + threadIdx.x;
  if (i < n) deg[i] = 0u;
}

// ---------------------------------------------------------------------------
// K1: in-degree of target nodes (self-loop added analytically later as +1)
// ---------------------------------------------------------------------------
__global__ void degree_kernel(const long long* __restrict__ ei,
                              unsigned* __restrict__ deg, int n_edges) {
  int e = blockIdx.x * blockDim.x + threadIdx.x;
  if (e < n_edges) {
    int c = (int)ei[n_edges + e];  // edge_index[1][e]
    __hip_atomic_fetch_add(&deg[c], 1u, __ATOMIC_RELAXED, __HIP_MEMORY_SCOPE_AGENT);
  }
}

// ---------------------------------------------------------------------------
// K2: dis[n] = deg^{-1/2} with self loop (deg+1 > 0 always)
// ---------------------------------------------------------------------------
__global__ void dis_kernel(const unsigned* __restrict__ deg,
                           float* __restrict__ dis, int n) {
  int i = blockIdx.x * blockDim.x + threadIdx.x;
  if (i < n) dis[i] = rsqrtf((float)(deg[i] + 1u));
}

// ---------------------------------------------------------------------------
// K3: fused fc1+ReLU -> WMMA GEMM (h1 @ gcn_w^T) -> scale by dis[row].
// One wave = one 16-node tile. A is built in-register straight from x (no
// h1 round trip), B fragments hold gcn_w^T. Writes g and seeds acc = g
// (the self-loop term of the scatter sum).
// ---------------------------------------------------------------------------
__global__ void __launch_bounds__(256)
fc1_gemm_kernel(const float* __restrict__ x,
                const float* __restrict__ fc1_w, const float* __restrict__ fc1_b,
                const float* __restrict__ gcn_w,
                const float* __restrict__ dis,
                float* __restrict__ g, float* __restrict__ acc, int n) {
  const int lane = threadIdx.x & 31;
  const int wave = threadIdx.x >> 5;
  const int row0 = (blockIdx.x * 8 + wave) * 16;  // first node of this tile
  const int kgrp = lane >> 4;                     // lane group (K-half select)
  const int mcol = lane & 15;

  // ---- A fragment (16x32 f16, M=node rows, K=features), matching B:
  // lanes 0-15 hold K in {0..15}; lanes 16-31 hold K in {16..31}.
  int arow = row0 + mcol;
  if (arow >= n) arow = n - 1;  // clamp loads; EXEC must stay all-ones for WMMA
  const float xv = x[arow];
  v16h a;
#pragma unroll
  for (int t = 0; t < 16; ++t) {
    int k = t + (kgrp << 4);
    float h1 = fmaxf(fmaf(xv, fc1_w[k], fc1_b[k]), 0.0f);  // fc1 + ReLU fused
    a[t] = (_Float16)h1;
  }

  // ---- B fragments (32x16 f16): B[k][col] = gcn_w[col][k]  (h2 = h1 @ W^T)
  // lane = output column, lane group selects K in [0,16) vs [16,32).
  const int kb = kgrp << 4;
  v16h b0, b1;
#pragma unroll
  for (int t = 0; t < 16; ++t) {
    b0[t] = (_Float16)gcn_w[mcol * F + kb + t];         // out cols 0..15
    b1[t] = (_Float16)gcn_w[(mcol + 16) * F + kb + t];  // out cols 16..31
  }

  v8f c0 = {}, c1 = {};
  c0 = __builtin_amdgcn_wmma_f32_16x16x32_f16(false, a, false, b0,
                                              (short)0, c0, false, false);
  c1 = __builtin_amdgcn_wmma_f32_16x16x32_f16(false, a, false, b1,
                                              (short)0, c1, false, false);

  // ---- C layout: VGPR v holds row v (lanes 0-15) / row v+8 (lanes 16-31),
  // column = lane&15. Scale by dis[row]; store g and seed acc with g.
  // N % 16 == 0 in practice, so nearly every tile takes the branch-free
  // fast path (no per-row exec-mask churn); only launch-overhang waves and
  // a ragged tail use the guarded path.
  const int mbase = kgrp << 3;
  if (row0 + 16 <= n) {
#pragma unroll
    for (int v = 0; v < 8; ++v) {
      const int r = row0 + mbase + v;
      const float dv = dis[r];
      const float g0 = c0[v] * dv;
      const float g1 = c1[v] * dv;
      g[r * F + mcol]        = g0;
      g[r * F + 16 + mcol]   = g1;
      acc[r * F + mcol]      = g0;  // self-loop contribution
      acc[r * F + 16 + mcol] = g1;
    }
  } else {
#pragma unroll
    for (int v = 0; v < 8; ++v) {
      const int r = row0 + mbase + v;
      if (r < n) {
        const float dv = dis[r];
        const float g0 = c0[v] * dv;
        const float g1 = c1[v] * dv;
        g[r * F + mcol]        = g0;
        g[r * F + 16 + mcol]   = g1;
        acc[r * F + mcol]      = g0;
        acc[r * F + 16 + mcol] = g1;
      }
    }
  }
}

// ---------------------------------------------------------------------------
// K4: edge scatter. acc[col] += g[row] -- no per-edge FLOPs (norm was hoisted
// into g and the final dis[col] multiply). 8 threads per edge so each edge's
// 128B gather is a coalesced wave access; native f32 fadd atomics hit L2
// (entire 25.6MB working set is resident in the 192MB L2).
// ---------------------------------------------------------------------------
__global__ void __launch_bounds__(256)
scatter_kernel(const long long* __restrict__ ei,
               const float* __restrict__ g, float* __restrict__ acc,
               int n_edges) {
  long long gid = (long long)blockIdx.x * blockDim.x + threadIdx.x;
  int e = (int)(gid >> 3);
  int q = ((int)gid & 7) << 2;  // float offset: 0,4,...,28
  if (e < n_edges) {
    int r = (int)ei[e];            // source
    int c = (int)ei[n_edges + e];  // target
    const float4 v = *(const float4*)(g + r * F + q);
    float* dst = acc + c * F + q;
    __hip_atomic_fetch_add(dst + 0, v.x, __ATOMIC_RELAXED, __HIP_MEMORY_SCOPE_AGENT);
    __hip_atomic_fetch_add(dst + 1, v.y, __ATOMIC_RELAXED, __HIP_MEMORY_SCOPE_AGENT);
    __hip_atomic_fetch_add(dst + 2, v.z, __ATOMIC_RELAXED, __HIP_MEMORY_SCOPE_AGENT);
    __hip_atomic_fetch_add(dst + 3, v.w, __ATOMIC_RELAXED, __HIP_MEMORY_SCOPE_AGENT);
  }
}

// ---------------------------------------------------------------------------
// K5: epilogue, one wave32 per node: out2 = dis*acc; ReLU(out2 + gcn_b);
// fc3 dot-product via 5-step butterfly reduction across the wave.
// ---------------------------------------------------------------------------
__global__ void __launch_bounds__(256)
finalize_kernel(const float* __restrict__ acc, const float* __restrict__ dis,
                const float* __restrict__ gcn_b, const float* __restrict__ fc3_w,
                const float* __restrict__ fc3_b, float* __restrict__ out, int n) {
  int node = blockIdx.x * 8 + (threadIdx.x >> 5);
  int lane = threadIdx.x & 31;
  if (node >= n) return;
  float h = fmaf(dis[node], acc[node * F + lane], gcn_b[lane]);
  h = fmaxf(h, 0.0f);
  float p = h * fc3_w[lane];
#pragma unroll
  for (int off = 16; off > 0; off >>= 1) p += __shfl_xor(p, off, 32);
  if (lane == 0) out[node] = p + fc3_b[0];
}

// ---------------------------------------------------------------------------
extern "C" void kernel_launch(void* const* d_in, const int* in_sizes, int n_in,
                              void* d_out, int out_size, void* d_ws, size_t ws_size,
                              hipStream_t stream) {
  const float*     x     = (const float*)d_in[0];
  const long long* ei    = (const long long*)d_in[1];  // int64 [2, E]
  const float*     fc1_w = (const float*)d_in[2];
  const float*     fc1_b = (const float*)d_in[3];
  const float*     gcn_w = (const float*)d_in[4];
  const float*     gcn_b = (const float*)d_in[5];
  const float*     fc3_w = (const float*)d_in[6];
  const float*     fc3_b = (const float*)d_in[7];
  float* out = (float*)d_out;

  const int n       = in_sizes[0];      // N nodes (x is [N,1])
  const int n_edges = in_sizes[1] / 2;  // E

  // Workspace carve-up, 256B aligned: deg | dis | g | acc  (~26.5 MB total)
  char* ws = (char*)d_ws;
  size_t off = 0;
  auto carve = [&](size_t bytes) -> void* {
    void* p = ws + off;
    off = (off + bytes + 255) & ~(size_t)255;
    return p;
  };
  unsigned* deg = (unsigned*)carve((size_t)n * sizeof(unsigned));
  float*    dis = (float*)   carve((size_t)n * sizeof(float));
  float*    g   = (float*)   carve((size_t)n * F * sizeof(float));
  float*    acc = (float*)   carve((size_t)n * F * sizeof(float));

  zero_deg_kernel<<<(n + 255) / 256, 256, 0, stream>>>(deg, n);
  degree_kernel<<<(n_edges + 255) / 256, 256, 0, stream>>>(ei, deg, n_edges);
  dis_kernel<<<(n + 255) / 256, 256, 0, stream>>>(deg, dis, n);

  const int ntiles = (n + 15) / 16;  // 16-node WMMA tiles, 8 waves per block
  fc1_gemm_kernel<<<(ntiles + 7) / 8, 256, 0, stream>>>(x, fc1_w, fc1_b, gcn_w,
                                                        dis, g, acc, n);

  const long long sthreads = (long long)n_edges * 8;
  scatter_kernel<<<(unsigned)((sthreads + 255) / 256), 256, 0, stream>>>(
      ei, g, acc, n_edges);

  finalize_kernel<<<(n + 7) / 8, 256, 0, stream>>>(acc, dis, gcn_b, fc3_w,
                                                   fc3_b, out, n);
}
